// MultiheadAttention_9689446220354
// MI455X (gfx1250) — matrix operand order repaired and checked by reading, not compiled, not run on hardware
//
#include <hip/hip_runtime.h>

// MultiheadAttention with post-softmax bias, MI455X (gfx1250, wave32, WMMA).
// Roofline: dist/path bias stream = 536 MB read-once -> ~23 us floor at
// 23.3 TB/s; all GEMM work (~34 GF bf16 incl. flash recompute) hides under it.
// Flash-style two-pass attention, never materializing [B,H,S,S] in HBM;
// all matmuls on v_wmma_f32_16x16x32_bf16; bias read once with NT hints +
// global_prefetch_b8 lookahead. GEMM k-steps load A + four named B fragments
// up front (in-order load returns -> partial s_wait_loadcnt) without the
// register cost of full double-buffering (which spilled to scratch).

#define BB      2
#define SS      2048
#define INDIM   512
#define EMBED   512
#define NH      8
#define HD      64
#define S3      (3 * EMBED)     // 1536
#define ROWS    (BB * SS)       // 4096

typedef __attribute__((ext_vector_type(16))) __bf16 v16bf;
typedef __attribute__((ext_vector_type(8)))  __bf16 v8bf;
typedef __attribute__((ext_vector_type(8)))  float  v8f;

__device__ __forceinline__ v8f zero8() {
    v8f z;
#pragma unroll
    for (int i = 0; i < 8; ++i) z[i] = 0.0f;
    return z;
}

// A fragment (16x32 bf16, row-major source, lda in elements).
// Lanes 0-15 hold M=lane with K=0..15;
// lanes 16-31 hold M=lane-16 with K=16..31.
__device__ __forceinline__ v16bf load_frag_a(const __bf16* A, int lda) {
    const int lane = threadIdx.x & 31;
    const int m    = lane & 15;
    const int kb   = (lane >> 4) << 4;          // 0 or 16
    const __bf16* p = A + m * lda + kb;
    v8bf lo = *reinterpret_cast<const v8bf*>(p);        // K = kb .. kb+7
    v8bf hi = *reinterpret_cast<const v8bf*>(p + 8);    // K = kb+8 .. kb+15
    v16bf f;
#pragma unroll
    for (int i = 0; i < 8; ++i) { f[i] = lo[i]; f[i + 8] = hi[i]; }
    return f;
}

// B fragment (32x16 bf16) from a K-contiguous ("Bt", N-major) buffer:
// Bt[n*ldb + k]. Lanes 0-15: N=lane, K=0..15; lanes 16-31: N=lane-16, K=16..31.
__device__ __forceinline__ v16bf load_frag_b(const __bf16* Bt, int ldb) {
    const int lane = threadIdx.x & 31;
    const int n    = lane & 15;
    const int kb   = (lane >> 4) << 4;          // 0 or 16
    return *reinterpret_cast<const v16bf*>(Bt + n * ldb + kb);
}

__device__ __forceinline__ v8f wmma_bf16(v16bf a, v16bf b, v8f c) {
    return __builtin_amdgcn_wmma_f32_16x16x32_bf16(
        false, a, false, b, (short)0, c, false, false);
}

// ---------------------------------------------------------------- converts --
__global__ void cvt_f32_bf16(const float* __restrict__ in, __bf16* __restrict__ out, int n) {
    int i = blockIdx.x * blockDim.x + threadIdx.x;
    if (i < n) out[i] = (__bf16)in[i];
}

// in: [K, N] f32 row-major -> out: [N, K] bf16
__global__ void cvt_transpose(const float* __restrict__ in, __bf16* __restrict__ out,
                              int K, int N) {
    int i = blockIdx.x * blockDim.x + threadIdx.x;
    if (i >= K * N) return;
    int n = i / K;
    int k = i - n * K;
    out[i] = (__bf16)in[k * N + n];
}

// ---------------------------------------------------------------- QKV GEMM --
// xb [ROWS, INDIM] bf16 x wT [S3, INDIM] bf16 (+bias) -> scatter q/k/vT.
// Each wave: 16x64 tile, A reused 4x, four named B fragments per k-step.
__global__ void qkv_gemm(const __bf16* __restrict__ xb, const __bf16* __restrict__ wT,
                         const float* __restrict__ bias,
                         __bf16* __restrict__ q, __bf16* __restrict__ k,
                         __bf16* __restrict__ vT) {
    const int wave = threadIdx.x >> 5;
    const int m0   = blockIdx.y << 4;
    const int n0   = (blockIdx.x << 8) + (wave << 6);   // 64 cols per wave
    const __bf16* arow = xb + (size_t)m0 * INDIM;
    const __bf16* brow = wT + (size_t)n0 * INDIM;
    v8f acc0 = zero8(), acc1 = zero8(), acc2 = zero8(), acc3 = zero8();

#pragma unroll 2
    for (int kk = 0; kk < INDIM; kk += 32) {
        v16bf a  = load_frag_a(arow + kk, INDIM);
        v16bf b0 = load_frag_b(brow + (size_t)(0 * 16) * INDIM + kk, INDIM);
        v16bf b1 = load_frag_b(brow + (size_t)(1 * 16) * INDIM + kk, INDIM);
        v16bf b2 = load_frag_b(brow + (size_t)(2 * 16) * INDIM + kk, INDIM);
        v16bf b3 = load_frag_b(brow + (size_t)(3 * 16) * INDIM + kk, INDIM);
        acc0 = wmma_bf16(a, b0, acc0);
        acc1 = wmma_bf16(a, b1, acc1);
        acc2 = wmma_bf16(a, b2, acc2);
        acc3 = wmma_bf16(a, b3, acc3);
    }

    const int lane = threadIdx.x & 31;
    const int nl   = lane & 15;
    const int mb   = (lane >> 4) << 3;
    v8f accs[4] = {acc0, acc1, acc2, acc3};
#pragma unroll
    for (int j = 0; j < 4; ++j) {
        const int n   = n0 + j * 16 + nl;
        const int h   = n / (3 * HD);
        const int rem = n - h * (3 * HD);
        const int t   = rem / HD;        // 0=q 1=k 2=v
        const int d   = rem - t * HD;
        const float bv = bias[n];
#pragma unroll
        for (int r = 0; r < 8; ++r) {
            int m    = m0 + mb + r;
            int bidx = m >> 11;          // / SS
            int s    = m & (SS - 1);
            float val = accs[j][r] + bv;
            if (t == 0)
                q[(((size_t)(bidx * NH + h) * SS) + s) * HD + d] = (__bf16)(val * 0.125f);
            else if (t == 1)
                k[(((size_t)(bidx * NH + h) * SS) + s) * HD + d] = (__bf16)val;
            else
                vT[(((size_t)(bidx * NH + h) * HD) + d) * SS + s] = (__bf16)val;
        }
    }
}

// --------------------------------------------------------------- attention --
// One block per (b*H+h, 16-query tile). 8 waves, each owns 16 key tiles.
// Pass 1: per-lane online softmax stats, ONE cross-lane (m,s) merge at end;
// K-tile fragments lightly double-buffered. Pass 2: recompute logits,
// softmax + NT-loaded dist/path bias, p->bf16 via LDS relayout, PV WMMA,
// cross-wave LDS reduce. Prefetch next bias chunk (global_prefetch_b8).
__global__ void __launch_bounds__(256, 2)
attn_kernel(const __bf16* __restrict__ q, const __bf16* __restrict__ kmat,
            const __bf16* __restrict__ vT,
            const float* __restrict__ dist, const float* __restrict__ path,
            __bf16* __restrict__ out /* [ROWS, EMBED] bf16 */) {
    const int bh   = blockIdx.y;           // b*NH + h
    const int q0   = blockIdx.x << 4;
    const int wave = threadIdx.x >> 5;
    const int lane = threadIdx.x & 31;
    const int nl   = lane & 15;
    const int mb   = (lane >> 4) << 3;

    const __bf16* Q = q    + (size_t)bh * SS * HD + (size_t)q0 * HD;
    const __bf16* K = kmat + (size_t)bh * SS * HD;
    const __bf16* V = vT   + (size_t)bh * HD * SS;
    const float*  D = dist + (size_t)bh * SS * SS + (size_t)q0 * SS;
    const float*  P = path + (size_t)bh * SS * SS + (size_t)q0 * SS;

    __shared__ __align__(32) float  stats_m[8][16];
    __shared__ __align__(32) float  stats_s[8][16];
    __shared__ __align__(32) float  rowM[16];
    __shared__ __align__(32) float  rowS[16];
    __shared__ __align__(32) __bf16 ptile[8][16 * 32];
    __shared__ __align__(32) float  vals[16][64];

    // q A-fragments: K = HD = 64 -> two 16x32 fragments
    v16bf qa0 = load_frag_a(Q, HD);
    v16bf qa1 = load_frag_a(Q + 32, HD);

    // ---- pass 1: per-lane online softmax statistics ----
    float run_m[8], run_s[8];
#pragma unroll
    for (int r = 0; r < 8; ++r) { run_m[r] = -3.0e38f; run_s[r] = 0.0f; }

    v16bf kb0 = load_frag_b(K + (size_t)wave * 16 * HD, HD);
    v16bf kb1 = load_frag_b(K + (size_t)wave * 16 * HD + 32, HD);
    for (int kt = wave; kt < SS / 16; kt += 8) {
        // double-buffer next key tile (clamped; harmless reload at the tail)
        const int ktn = (kt + 8 < SS / 16) ? (kt + 8) : kt;
        v16bf nb0 = load_frag_b(K + (size_t)ktn * 16 * HD, HD);
        v16bf nb1 = load_frag_b(K + (size_t)ktn * 16 * HD + 32, HD);
        v8f c = zero8();
        c = wmma_bf16(qa0, kb0, c);
        c = wmma_bf16(qa1, kb1, c);
#pragma unroll
        for (int r = 0; r < 8; ++r) {
            float x  = c[r];
            float nm = fmaxf(run_m[r], x);
            run_s[r] = run_s[r] * __expf(run_m[r] - nm) + __expf(x - nm);
            run_m[r] = nm;
        }
        kb0 = nb0; kb1 = nb1;
    }
    // single cross-lane (m,s) merge over the 16-lane half-groups
#pragma unroll
    for (int r = 0; r < 8; ++r) {
#pragma unroll
        for (int off = 1; off < 16; off <<= 1) {
            float om = __shfl_xor(run_m[r], off, 32);
            float os = __shfl_xor(run_s[r], off, 32);
            float nm = fmaxf(run_m[r], om);
            run_s[r] = run_s[r] * __expf(run_m[r] - nm) + os * __expf(om - nm);
            run_m[r] = nm;
        }
    }
    if (nl == 0) {
#pragma unroll
        for (int r = 0; r < 8; ++r) {
            stats_m[wave][mb + r] = run_m[r];
            stats_s[wave][mb + r] = run_s[r];
        }
    }
    __syncthreads();
    if (threadIdx.x < 16) {
        float gm = -3.0e38f;
        for (int w = 0; w < 8; ++w) gm = fmaxf(gm, stats_m[w][threadIdx.x]);
        float gs = 0.0f;
        for (int w = 0; w < 8; ++w)
            gs += stats_s[w][threadIdx.x] * __expf(stats_m[w][threadIdx.x] - gm);
        rowM[threadIdx.x] = gm;
        rowS[threadIdx.x] = 1.0f / gs;
    }
    for (int i = threadIdx.x; i < 16 * 64; i += 256) ((float*)vals)[i] = 0.0f;
    __syncthreads();

    float Mrow[8], Sinv[8];
#pragma unroll
    for (int r = 0; r < 8; ++r) { Mrow[r] = rowM[mb + r]; Sinv[r] = rowS[mb + r]; }

    // ---- pass 2: recompute, bias, PV ----
    v8f acc_o[4];
#pragma unroll
    for (int nt = 0; nt < 4; ++nt) acc_o[nt] = zero8();

    __bf16* pt = &ptile[wave][0];
    for (int kc = wave * 32; kc < SS; kc += 8 * 32) {
        // prefetch next chunk's bias rows (lanes 0-15: dist, 16-31: path)
        {
            const int kc2 = kc + 8 * 32;
            if (kc2 < SS) {
                const float* pre = (lane < 16) ? D : P;
                __builtin_prefetch(pre + (size_t)nl * SS + kc2, 0, 1);
            }
        }
#pragma unroll
        for (int half = 0; half < 2; ++half) {
            const int kt = kc + half * 16;
            const __bf16* Kt = K + (size_t)kt * HD;
            v16bf b0 = load_frag_b(Kt, HD);
            v16bf b1 = load_frag_b(Kt + 32, HD);
            v8f c = zero8();
            c = wmma_bf16(qa0, b0, c);
            c = wmma_bf16(qa1, b1, c);
            const int col = kt + nl;
#pragma unroll
            for (int r = 0; r < 8; ++r) {
                const int row = mb + r;
                float dv = __builtin_nontemporal_load(&D[row * SS + col]);
                float pv = __builtin_nontemporal_load(&P[row * SS + col]);
                float pr = __expf(c[r] - Mrow[r]) * Sinv[r] + dv + pv;
                pt[row * 32 + half * 16 + nl] = (__bf16)pr;
            }
        }
        // p tile 16x32 (LDS) as A-fragment; V rows kc..kc+31 as B-fragments
        v16bf pa = load_frag_a(pt, 32);
        v16bf vb0 = load_frag_b(V + (size_t)(0 * 16) * SS + kc, SS);
        v16bf vb1 = load_frag_b(V + (size_t)(1 * 16) * SS + kc, SS);
        v16bf vb2 = load_frag_b(V + (size_t)(2 * 16) * SS + kc, SS);
        v16bf vb3 = load_frag_b(V + (size_t)(3 * 16) * SS + kc, SS);
        acc_o[0] = wmma_bf16(pa, vb0, acc_o[0]);
        acc_o[1] = wmma_bf16(pa, vb1, acc_o[1]);
        acc_o[2] = wmma_bf16(pa, vb2, acc_o[2]);
        acc_o[3] = wmma_bf16(pa, vb3, acc_o[3]);
    }
    // cross-wave reduction (ds_add_f32)
#pragma unroll
    for (int nt = 0; nt < 4; ++nt)
#pragma unroll
        for (int r = 0; r < 8; ++r)
            atomicAdd(&vals[mb + r][nt * 16 + nl], acc_o[nt][r]);
    __syncthreads();

    const int h    = bh & (NH - 1);
    const int bidx = bh >> 3;
    for (int i = threadIdx.x; i < 16 * 64; i += 256) {
        const int row = i >> 6, d = i & 63;
        const size_t m = (size_t)bidx * SS + q0 + row;
        out[m * EMBED + h * HD + d] = (__bf16)vals[row][d];
    }
}

// ----------------------------------------------------------- out projection --
// Each wave: 16x64 tile, A reused 4x, four named B fragments per k-step.
__global__ void out_gemm(const __bf16* __restrict__ vb, const __bf16* __restrict__ woT,
                         const float* __restrict__ bias, float* __restrict__ out) {
    const int wave = threadIdx.x >> 5;
    const int m0   = blockIdx.y << 4;
    const int n0   = (blockIdx.x << 8) + (wave << 6);
    const __bf16* arow = vb  + (size_t)m0 * EMBED;
    const __bf16* brow = woT + (size_t)n0 * EMBED;
    v8f acc0 = zero8(), acc1 = zero8(), acc2 = zero8(), acc3 = zero8();

#pragma unroll 2
    for (int kk = 0; kk < EMBED; kk += 32) {
        v16bf a  = load_frag_a(arow + kk, EMBED);
        v16bf b0 = load_frag_b(brow + (size_t)(0 * 16) * EMBED + kk, EMBED);
        v16bf b1 = load_frag_b(brow + (size_t)(1 * 16) * EMBED + kk, EMBED);
        v16bf b2 = load_frag_b(brow + (size_t)(2 * 16) * EMBED + kk, EMBED);
        v16bf b3 = load_frag_b(brow + (size_t)(3 * 16) * EMBED + kk, EMBED);
        acc0 = wmma_bf16(a, b0, acc0);
        acc1 = wmma_bf16(a, b1, acc1);
        acc2 = wmma_bf16(a, b2, acc2);
        acc3 = wmma_bf16(a, b3, acc3);
    }

    const int lane = threadIdx.x & 31;
    const int nl   = lane & 15;
    const int mb   = (lane >> 4) << 3;
    v8f accs[4] = {acc0, acc1, acc2, acc3};
#pragma unroll
    for (int j = 0; j < 4; ++j) {
        const float bv = bias[n0 + j * 16 + nl];
#pragma unroll
        for (int r = 0; r < 8; ++r)
            out[(size_t)(m0 + mb + r) * EMBED + n0 + j * 16 + nl] = accs[j][r] + bv;
    }
}

// ------------------------------------------------------------------- launch --
extern "C" void kernel_launch(void* const* d_in, const int* in_sizes, int n_in,
                              void* d_out, int out_size, void* d_ws, size_t ws_size,
                              hipStream_t stream) {
    const float* x      = (const float*)d_in[0];
    const float* dist   = (const float*)d_in[1];
    const float* path   = (const float*)d_in[2];
    const float* qkv_w  = (const float*)d_in[3];
    const float* qkv_b  = (const float*)d_in[4];
    const float* o_w    = (const float*)d_in[5];
    const float* o_b    = (const float*)d_in[6];
    float* out = (float*)d_out;

    char* ws = (char*)d_ws;
    size_t off = 0;
    __bf16* xb    = (__bf16*)(ws + off); off += (size_t)ROWS * INDIM * 2;   // 4 MB
    __bf16* wqkvT = (__bf16*)(ws + off); off += (size_t)S3 * INDIM * 2;     // 1.5 MB
    __bf16* woT   = (__bf16*)(ws + off); off += (size_t)EMBED * EMBED * 2;  // 0.5 MB
    __bf16* qbuf  = (__bf16*)(ws + off); off += (size_t)BB * NH * SS * HD * 2;
    __bf16* kbuf  = (__bf16*)(ws + off); off += (size_t)BB * NH * SS * HD * 2;
    __bf16* vTbuf = (__bf16*)(ws + off); off += (size_t)BB * NH * HD * SS * 2;
    __bf16* valsb = (__bf16*)(ws + off); off += (size_t)ROWS * EMBED * 2;
    (void)off; (void)ws_size; (void)in_sizes; (void)n_in; (void)out_size;

    cvt_f32_bf16<<<(ROWS * INDIM + 255) / 256, 256, 0, stream>>>(x, xb, ROWS * INDIM);
    cvt_transpose<<<(S3 * INDIM + 255) / 256, 256, 0, stream>>>(qkv_w, wqkvT, INDIM, S3);
    cvt_transpose<<<(EMBED * EMBED + 255) / 256, 256, 0, stream>>>(o_w, woT, EMBED, EMBED);

    qkv_gemm<<<dim3(S3 / 256, ROWS / 16), 128, 0, stream>>>(xb, wqkvT, qkv_b,
                                                            qbuf, kbuf, vTbuf);

    attn_kernel<<<dim3(SS / 16, BB * NH), 256, 0, stream>>>(qbuf, kbuf, vTbuf,
                                                            dist, path, valsb);

    out_gemm<<<dim3(EMBED / 256, ROWS / 16), 128, 0, stream>>>(valsb, woT, o_b, out);
}
